// SAMTransformerPromptGenNeck_584115552530
// MI455X (gfx1250) — compile-verified, matrix operand order checked by reading, not run on hardware
//
#include <hip/hip_runtime.h>
#include <math.h>

typedef __attribute__((ext_vector_type(16))) _Float16 v16h;
typedef __attribute__((ext_vector_type(8)))  _Float16 v8h;
typedef __attribute__((ext_vector_type(8)))  float    v8f;
typedef __attribute__((ext_vector_type(4)))  float    f4;

#define NQ 4096
#define NP 600
#define EMB_ 256

// Load 8 contiguous f32 from global (2x b128), convert to 8 halves
__device__ __forceinline__ v8h cvt8(const float* __restrict__ g) {
  f4 a = *(const f4*)g;
  f4 b = *(const f4*)(g + 4);
  v8h h;
  h[0] = (_Float16)a[0]; h[1] = (_Float16)a[1]; h[2] = (_Float16)a[2]; h[3] = (_Float16)a[3];
  h[4] = (_Float16)b[0]; h[5] = (_Float16)b[1]; h[6] = (_Float16)b[2]; h[7] = (_Float16)b[3];
  return h;
}

// Build a 16-half fragment from two aligned 8-half LDS groups (2x ds_load_b128)
__device__ __forceinline__ v16h frag16(const _Float16* p0, const _Float16* p1) {
  v8h lo = *(const v8h*)p0;
  v8h hi = *(const v8h*)p1;
  return __builtin_shufflevector(lo, hi, 0, 1, 2, 3, 4, 5, 6, 7, 8, 9, 10, 11, 12, 13, 14, 15);
}

// ---------------------------------------------------------------------------
// Transpose [C=256, T=4096] -> [T, C]
// ---------------------------------------------------------------------------
__global__ __launch_bounds__(256) void transpose_ct(const float* __restrict__ in,
                                                    float* __restrict__ out) {
  __shared__ float tile[32][33];
  const int bx = blockIdx.x * 32;            // t
  const int by = blockIdx.y * 32;            // c
  const int tx = threadIdx.x & 31;
  const int ty = threadIdx.x >> 5;           // 0..7
  #pragma unroll
  for (int i = 0; i < 32; i += 8)
    tile[ty + i][tx] = in[(by + ty + i) * 4096 + bx + tx];
  __syncthreads();
  #pragma unroll
  for (int i = 0; i < 32; i += 8)
    out[(bx + ty + i) * 256 + by + tx] = tile[tx][ty + i];
}

// ---------------------------------------------------------------------------
// Elementwise add
// ---------------------------------------------------------------------------
__global__ __launch_bounds__(256) void addv(const float* __restrict__ a,
                                            const float* __restrict__ b,
                                            float* __restrict__ o, int n) {
  int i = blockIdx.x * 256 + threadIdx.x;
  if (i < n) o[i] = a[i] + b[i];
}

// ---------------------------------------------------------------------------
// WMMA GEMM: C[M,N] = A[M,K] @ B[K,N] + bias (+Res / GELU)
// block = 256 threads = 8 waves; block tile 64x64, wave tile 16x32, BK=32
// Requires: K % 32 == 0, N % 64 == 0 (true for all call sites); M arbitrary.
// ---------------------------------------------------------------------------
enum { EPI_NONE = 0, EPI_RES = 1, EPI_GELU = 2 };

template <int EPI>
__global__ __launch_bounds__(256) void gemm_wmma(const float* __restrict__ A,
                                                 const float* __restrict__ B,
                                                 const float* __restrict__ bias,
                                                 const float* __restrict__ Res,
                                                 float* __restrict__ C,
                                                 int M, int N, int K) {
  __shared__ _Float16 Ah[64][40];   // row pitch 80B (16B multiple)
  __shared__ _Float16 Bt[64][40];   // B transposed: Bt[n][k]
  const int tid  = threadIdx.x;
  const int wave = tid >> 5;
  const int lane = tid & 31;
  const int wrow = wave >> 1;          // 0..3
  const int wcol = wave & 1;           // 0..1
  const int l15  = lane & 15;
  const int khalf = (lane >> 4) * 8;
  const int bm0 = blockIdx.x * 64;
  const int bn0 = blockIdx.y * 64;

  // staging coordinates (group-uniform guards)
  const int ar  = tid >> 2;            // A row 0..63
  const int ac8 = (tid & 3) * 8;       // A col group
  const int bkr = tid >> 3;            // B row (k) 0..31
  const int bn8 = (tid & 7) * 8;       // B col group 0..56

  v8f acc0 = {}; v8f acc1 = {};

  for (int k0 = 0; k0 < K; k0 += 32) {
    { // stage A tile 64x32: one v8h per thread
      int gm = bm0 + ar;
      v8h hv = {};
      if (gm < M) hv = cvt8(&A[(size_t)gm * K + k0 + ac8]);
      *(v8h*)&Ah[ar][ac8] = hv;
    }
    { // stage B tile 32x64, transposed into Bt[n][k]
      v8h hv = cvt8(&B[(size_t)(k0 + bkr) * N + bn0 + bn8]);
      #pragma unroll
      for (int i = 0; i < 8; ++i) Bt[bn8 + i][bkr] = hv[i];
    }
    __syncthreads();
    const int m_loc = wrow * 16 + l15;
    const int n0 = wcol * 32 + l15;
    v16h a  = frag16(&Ah[m_loc][khalf],   &Ah[m_loc][16 + khalf]);
    v16h b0 = frag16(&Bt[n0][khalf],      &Bt[n0][16 + khalf]);
    v16h b1 = frag16(&Bt[n0 + 16][khalf], &Bt[n0 + 16][16 + khalf]);
    acc0 = __builtin_amdgcn_wmma_f32_16x16x32_f16(false, a, false, b0, (short)0, acc0, false, false);
    acc1 = __builtin_amdgcn_wmma_f32_16x16x32_f16(false, a, false, b1, (short)0, acc1, false, false);
    __syncthreads();
  }

  float accf[2][8];
  #pragma unroll
  for (int r = 0; r < 8; ++r) { accf[0][r] = acc0[r]; accf[1][r] = acc1[r]; }
  const int mbase = bm0 + wrow * 16 + (lane >> 4) * 8;
  #pragma unroll
  for (int s = 0; s < 2; ++s) {
    int n = bn0 + wcol * 32 + s * 16 + l15;
    float bv = bias ? bias[n] : 0.f;
    #pragma unroll
    for (int r = 0; r < 8; ++r) {
      int m = mbase + r;
      if (m < M) {
        float v = accf[s][r] + bv;
        if (EPI == EPI_GELU) v = 0.5f * v * (1.f + erff(v * 0.70710678118654752f));
        if (EPI == EPI_RES)  v += Res[(size_t)m * N + n];
        C[(size_t)m * N + n] = v;
      }
    }
  }
}

// ---------------------------------------------------------------------------
// Fused attention: one block = 16 queries of one head. Scores live in LDS.
//   S = softmax(Q Kh^T * scale); O = S V  (per head slice of width HD)
// LDS: Qs 16x32 f16 | KV stage (K rows 128x32 / V^T 32x264) | S 16x(NKMAX+8) f16
// ---------------------------------------------------------------------------
#define KVHALVES 8448   // max(128*32, 32*264)
#define VTP 264         // V^T row pitch in halves (528B, 16B multiple)

__host__ __device__ constexpr size_t attn_smem_bytes(int NKMAX) {
  return 256 * 4          // red
       + 16 * 4           // rowmax
       + 16 * 4           // rowsum
       + 16 * 32 * 4      // Oacc
       + 16 * 32 * 2      // Qs
       + KVHALVES * 2     // KV stage
       + (size_t)16 * (NKMAX + 8) * 2;  // S (f16)
}

template <int HD, int NKMAX>
__global__ __launch_bounds__(256) void attn_wmma(const float* __restrict__ Q,
                                                 const float* __restrict__ Kmat,
                                                 const float* __restrict__ V,
                                                 float* __restrict__ O,
                                                 int Nq, int Nk, int di, float scale) {
  extern __shared__ char smem[];
  float*    red    = (float*)smem;
  float*    rowmax = red + 256;
  float*    rowsum = rowmax + 16;
  float*    Oacc   = rowsum + 16;
  _Float16* Qs     = (_Float16*)(Oacc + 16 * 32);
  _Float16* KV     = Qs + 16 * 32;
  _Float16* S      = KV + KVHALVES;
  const int NKP = NKMAX + 8;

  const int tid  = threadIdx.x;
  const int wave = tid >> 5;
  const int lane = tid & 31;
  const int l15  = lane & 15;
  const int khalf = (lane >> 4) * 8;
  const int h     = blockIdx.y;
  const int qbase = blockIdx.x * 16;
  const int hoff  = h * HD;
  const int NK32  = (Nk + 31) & ~31;

  // ---- stage Q tile (16 x 32, zero-padded), one v8h per thread ----
  if (tid < 64) {
    int r = tid >> 2, c8 = (tid & 3) * 8;
    int gm = qbase + r;
    v8h hv = {};
    if (gm < Nq && c8 < HD) hv = cvt8(&Q[(size_t)gm * di + hoff + c8]);
    *(v8h*)&Qs[r * 32 + c8] = hv;
  }
  __syncthreads();
  v16h aq = frag16(&Qs[l15 * 32 + khalf], &Qs[l15 * 32 + 16 + khalf]);

  // ---- phase 1: scores S = Q Kh^T * scale ----
  for (int c0 = 0; c0 < Nk; c0 += 128) {
    #pragma unroll
    for (int t = 0; t < 2; ++t) {           // stage K rows c0..c0+127 (128x32)
      int g = tid + t * 256;
      int r = g >> 2, c8 = (g & 3) * 8;
      int gk = c0 + r;
      v8h hv = {};
      if (gk < Nk && c8 < HD) hv = cvt8(&Kmat[(size_t)gk * di + hoff + c8]);
      *(v8h*)&KV[r * 32 + c8] = hv;
    }
    __syncthreads();
    const int nl = wave * 16 + l15;
    v16h bk = frag16(&KV[nl * 32 + khalf], &KV[nl * 32 + 16 + khalf]);
    v8f sc = {};
    sc = __builtin_amdgcn_wmma_f32_16x16x32_f16(false, aq, false, bk, (short)0, sc, false, false);
    const int gn = c0 + wave * 16 + l15;
    if (gn < Nk) {
      #pragma unroll
      for (int r = 0; r < 8; ++r)
        S[((lane >> 4) * 8 + r) * NKP + gn] = (_Float16)(sc[r] * scale);
    }
    __syncthreads();
  }

  // ---- phase 2: row softmax in LDS (group g of 16 threads owns row g) ----
  {
    const int g = tid >> 4, j0 = tid & 15;
    const int NkV = Nk & ~7;                // vectorizable portion (8-aligned)
    float lmax = -3.4e38f;
    for (int j = j0 * 8; j < NkV; j += 128) {
      v8h s8 = *(const v8h*)&S[g * NKP + j];
      #pragma unroll
      for (int e = 0; e < 8; ++e) lmax = fmaxf(lmax, (float)s8[e]);
    }
    for (int j = NkV + j0; j < Nk; j += 16) lmax = fmaxf(lmax, (float)S[g * NKP + j]);
    red[g * 16 + j0] = lmax;
    __syncthreads();
    if (j0 == 0) {
      float mx = red[g * 16];
      for (int t = 1; t < 16; ++t) mx = fmaxf(mx, red[g * 16 + t]);
      rowmax[g] = mx;
    }
    __syncthreads();
    const float mx = rowmax[g];
    float lsum = 0.f;
    for (int j = j0 * 8; j < NkV; j += 128) {
      v8h s8 = *(const v8h*)&S[g * NKP + j];
      #pragma unroll
      for (int e = 0; e < 8; ++e) {
        float p = expf((float)s8[e] - mx);
        s8[e] = (_Float16)p;
        lsum += p;
      }
      *(v8h*)&S[g * NKP + j] = s8;
    }
    for (int j = NkV + j0; j < Nk; j += 16) {
      float p = expf((float)S[g * NKP + j] - mx);
      S[g * NKP + j] = (_Float16)p;
      lsum += p;
    }
    for (int j = Nk + j0; j < NK32; j += 16) S[g * NKP + j] = (_Float16)0.f;
    red[g * 16 + j0] = lsum;
    __syncthreads();
    if (j0 == 0) {
      float s = 0.f;
      for (int t = 0; t < 16; ++t) s += red[g * 16 + t];
      rowsum[g] = s;
    }
  }
  for (int i = tid; i < 16 * HD; i += 256) Oacc[i] = 0.f;
  __syncthreads();

  // ---- phase 3: O = P V; V staged transposed (Vt[n][k]); Nk split by wave ----
  constexpr int NT = HD / 16;
  v8f oa0 = {}; v8f oa1 = {};
  for (int c0 = 0; c0 < NK32; c0 += 256) {
    #pragma unroll
    for (int t = 0; t < 4; ++t) {          // stage V rows c0..c0+255 transposed
      int g = tid + t * 256;
      int r = g >> 2, c8 = (g & 3) * 8;    // r: 0..255 (k), c8: col group
      int gk = c0 + r;
      v8h hv = {};
      if (gk < Nk && c8 < HD) hv = cvt8(&V[(size_t)gk * di + hoff + c8]);
      #pragma unroll
      for (int i = 0; i < 8; ++i) KV[(c8 + i) * VTP + r] = hv[i];
    }
    __syncthreads();
    const int k0 = wave * 32;
    if (c0 + k0 < NK32) {
      v16h pa  = frag16(&S[l15 * NKP + c0 + k0 + khalf],
                        &S[l15 * NKP + c0 + k0 + 16 + khalf]);
      v16h vb0 = frag16(&KV[l15 * VTP + k0 + khalf],
                        &KV[l15 * VTP + k0 + 16 + khalf]);
      oa0 = __builtin_amdgcn_wmma_f32_16x16x32_f16(false, pa, false, vb0, (short)0, oa0, false, false);
      if (NT == 2) {
        v16h vb1 = frag16(&KV[(16 + l15) * VTP + k0 + khalf],
                          &KV[(16 + l15) * VTP + k0 + 16 + khalf]);
        oa1 = __builtin_amdgcn_wmma_f32_16x16x32_f16(false, pa, false, vb1, (short)0, oa1, false, false);
      }
    }
    __syncthreads();
  }
  float of0[8], of1[8];
  #pragma unroll
  for (int r = 0; r < 8; ++r) { of0[r] = oa0[r]; of1[r] = oa1[r]; }
  #pragma unroll
  for (int r = 0; r < 8; ++r) {
    atomicAdd(&Oacc[((lane >> 4) * 8 + r) * HD + l15], of0[r]);
    if (NT == 2) atomicAdd(&Oacc[((lane >> 4) * 8 + r) * HD + 16 + l15], of1[r]);
  }
  __syncthreads();
  for (int i = tid; i < 16 * HD; i += 256) {
    int m = i / HD, n = i % HD;
    int gm = qbase + m;
    if (gm < Nq) O[(size_t)gm * di + hoff + n] = Oacc[i] / rowsum[m];
  }
}

// ---------------------------------------------------------------------------
// LayerNorm over last dim (256), one block per row
// ---------------------------------------------------------------------------
__global__ __launch_bounds__(256) void layernorm256(const float* __restrict__ X,
                                                    const float* __restrict__ g,
                                                    const float* __restrict__ b,
                                                    float* __restrict__ Y) {
  __shared__ float red[256];
  const int row = blockIdx.x, t = threadIdx.x;
  float v = X[(size_t)row * 256 + t];
  red[t] = v;
  __syncthreads();
  for (int s = 128; s > 0; s >>= 1) { if (t < s) red[t] += red[t + s]; __syncthreads(); }
  float mu = red[0] * (1.f / 256.f);
  __syncthreads();
  float d = v - mu;
  red[t] = d * d;
  __syncthreads();
  for (int s = 128; s > 0; s >>= 1) { if (t < s) red[t] += red[t + s]; __syncthreads(); }
  float var = red[0] * (1.f / 256.f);
  Y[(size_t)row * 256 + t] = d * rsqrtf(var + 1e-5f) * g[t] + b[t];
}

// ---------------------------------------------------------------------------
extern "C" void kernel_launch(void* const* d_in, const int* in_sizes, int n_in,
                              void* d_out, int out_size, void* d_ws, size_t ws_size,
                              hipStream_t stream) {
  (void)in_sizes; (void)n_in; (void)out_size; (void)ws_size;
  const float* img   = (const float*)d_in[0];
  const float* imgpe = (const float*)d_in[1];
  const float* pts   = (const float*)d_in[2];
  const float* ptspe = (const float*)d_in[3];

  struct AttnP { const float *wq,*bq,*wk,*bk,*wv,*bv,*wo,*bo; };
  int idx = 4;
  auto getA = [&]() { AttnP p;
    p.wq=(const float*)d_in[idx++]; p.bq=(const float*)d_in[idx++];
    p.wk=(const float*)d_in[idx++]; p.bk=(const float*)d_in[idx++];
    p.wv=(const float*)d_in[idx++]; p.bv=(const float*)d_in[idx++];
    p.wo=(const float*)d_in[idx++]; p.bo=(const float*)d_in[idx++];
    return p; };
  AttnP sa = getA(), ca1 = getA(), ca2 = getA(), fa = getA();
  const float* ln1g=(const float*)d_in[idx++]; const float* ln1b=(const float*)d_in[idx++];
  const float* ln2g=(const float*)d_in[idx++]; const float* ln2b=(const float*)d_in[idx++];
  const float* ln3g=(const float*)d_in[idx++]; const float* ln3b=(const float*)d_in[idx++];
  const float* ln4g=(const float*)d_in[idx++]; const float* ln4b=(const float*)d_in[idx++];
  const float* lnfg=(const float*)d_in[idx++]; const float* lnfb=(const float*)d_in[idx++];
  const float* w1=(const float*)d_in[idx++]; const float* b1=(const float*)d_in[idx++];
  const float* w2=(const float*)d_in[idx++]; const float* b2=(const float*)d_in[idx++];

  // workspace carve
  float* ws = (float*)d_ws;
  size_t off = 0;
  auto walloc = [&](size_t n) { float* p = ws + off; off += n; return p; };
  float* XQ  = walloc((size_t)NQ * EMB_);
  float* XPE = walloc((size_t)NQ * EMB_);
  float* TQ  = walloc((size_t)NQ * EMB_);
  float* TK  = walloc((size_t)NP * EMB_);
  float* Qb  = walloc((size_t)NQ * EMB_);
  float* Kb  = walloc((size_t)NQ * EMB_);
  float* Vb  = walloc((size_t)NQ * EMB_);
  float* Ob  = walloc((size_t)NQ * EMB_);
  float* Xr  = walloc((size_t)NQ * EMB_);
  float* Xn  = walloc((size_t)NQ * EMB_);
  float* Hid = walloc((size_t)NQ * 1024);
  float* Pr  = walloc((size_t)NP * EMB_);

  float* outQ = (float*)d_out;
  float* outK = outQ + (size_t)NQ * EMB_;

  auto gemm = [&](int epi, const float* A, const float* B, const float* bias,
                  const float* Res, float* C, int M, int N, int K) {
    dim3 g((M + 63) / 64, (N + 63) / 64);
    if (epi == EPI_NONE)      gemm_wmma<EPI_NONE><<<g, 256, 0, stream>>>(A, B, bias, Res, C, M, N, K);
    else if (epi == EPI_RES)  gemm_wmma<EPI_RES><<<g, 256, 0, stream>>>(A, B, bias, Res, C, M, N, K);
    else                      gemm_wmma<EPI_GELU><<<g, 256, 0, stream>>>(A, B, bias, Res, C, M, N, K);
  };
  const size_t smBig   = attn_smem_bytes(4096);
  const size_t smSmall = attn_smem_bytes(640);
  const float inv_sqrt32 = 0.17677669529663687f;

  // img [1,256,64,64] -> [4096,256]
  transpose_ct<<<dim3(128, 8), 256, 0, stream>>>(img, XQ);
  transpose_ct<<<dim3(128, 8), 256, 0, stream>>>(imgpe, XPE);

  // (1) self-attention of image tokens (di=256, hd=32)
  addv<<<(NQ * EMB_ + 255) / 256, 256, 0, stream>>>(XQ, XPE, TQ, NQ * EMB_);
  gemm(EPI_NONE, TQ, sa.wq, sa.bq, nullptr, Qb, NQ, 256, 256);
  gemm(EPI_NONE, TQ, sa.wk, sa.bk, nullptr, Kb, NQ, 256, 256);
  gemm(EPI_NONE, XQ, sa.wv, sa.bv, nullptr, Vb, NQ, 256, 256);
  attn_wmma<32, 4096><<<dim3(NQ / 16, 8), 256, smBig, stream>>>(Qb, Kb, Vb, Ob, NQ, NQ, 256, inv_sqrt32);
  gemm(EPI_RES, Ob, sa.wo, sa.bo, XQ, Xr, NQ, 256, 256);
  layernorm256<<<NQ, 256, 0, stream>>>(Xr, ln1g, ln1b, Xn);

  // (2) cross attention token->image (di=128, hd=16)
  addv<<<(NQ * EMB_ + 255) / 256, 256, 0, stream>>>(Xn, XPE, TQ, NQ * EMB_);
  addv<<<(NP * EMB_ + 255) / 256, 256, 0, stream>>>(pts, ptspe, TK, NP * EMB_);
  gemm(EPI_NONE, TQ, ca1.wq, ca1.bq, nullptr, Qb, NQ, 128, 256);
  gemm(EPI_NONE, TK, ca1.wk, ca1.bk, nullptr, Kb, NP, 128, 256);
  gemm(EPI_NONE, pts, ca1.wv, ca1.bv, nullptr, Vb, NP, 128, 256);
  attn_wmma<16, 640><<<dim3(NQ / 16, 8), 256, smSmall, stream>>>(Qb, Kb, Vb, Ob, NQ, NP, 128, 0.25f);
  gemm(EPI_RES, Ob, ca1.wo, ca1.bo, Xn, Xr, NQ, 256, 128);
  layernorm256<<<NQ, 256, 0, stream>>>(Xr, ln2g, ln2b, Xn);

  // (3) MLP with exact GELU
  gemm(EPI_GELU, Xn, w1, b1, nullptr, Hid, NQ, 1024, 256);
  gemm(EPI_RES, Hid, w2, b2, Xn, Xr, NQ, 256, 1024);
  layernorm256<<<NQ, 256, 0, stream>>>(Xr, ln3g, ln3b, Xn);

  // (4) cross attention image->token (point tokens are queries)
  addv<<<(NQ * EMB_ + 255) / 256, 256, 0, stream>>>(Xn, XPE, TQ, NQ * EMB_);
  addv<<<(NP * EMB_ + 255) / 256, 256, 0, stream>>>(pts, ptspe, TK, NP * EMB_);
  gemm(EPI_NONE, TK, ca2.wq, ca2.bq, nullptr, Qb, NP, 128, 256);
  gemm(EPI_NONE, TQ, ca2.wk, ca2.bk, nullptr, Kb, NQ, 128, 256);
  gemm(EPI_NONE, Xn, ca2.wv, ca2.bv, nullptr, Vb, NQ, 128, 256);
  attn_wmma<16, 4096><<<dim3((NP + 15) / 16, 8), 256, smBig, stream>>>(Qb, Kb, Vb, Ob, NP, NQ, 128, 0.25f);
  gemm(EPI_RES, Ob, ca2.wo, ca2.bo, pts, Pr, NP, 256, 128);
  layernorm256<<<NP, 256, 0, stream>>>(Pr, ln4g, ln4b, outK);   // keys output

  // (5) final attention token->image (key PE = original point_embedding)
  // TQ still holds Xn + XPE (unchanged since step 4)
  addv<<<(NP * EMB_ + 255) / 256, 256, 0, stream>>>(outK, pts, TK, NP * EMB_);
  gemm(EPI_NONE, TQ, fa.wq, fa.bq, nullptr, Qb, NQ, 128, 256);
  gemm(EPI_NONE, TK, fa.wk, fa.bk, nullptr, Kb, NP, 128, 256);
  gemm(EPI_NONE, outK, fa.wv, fa.bv, nullptr, Vb, NP, 128, 256);
  attn_wmma<16, 640><<<dim3(NQ / 16, 8), 256, smSmall, stream>>>(Qb, Kb, Vb, Ob, NQ, NP, 128, 0.25f);
  gemm(EPI_RES, Ob, fa.wo, fa.bo, Xn, Xr, NQ, 256, 128);
  layernorm256<<<NQ, 256, 0, stream>>>(Xr, lnfg, lnfb, outQ);   // queries output
}
